// SelfAttention_26714696581764
// MI455X (gfx1250) — matrix operand order repaired and checked by reading, not compiled, not run on hardware
//
#include <hip/hip_runtime.h>
#include <hip/hip_bf16.h>

// ---------------------------------------------------------------------------
// Problem constants (from reference): x (2,128,16,16,16) f32
// ---------------------------------------------------------------------------
#define BATCH   2
#define CCH     128
#define LTOK    4096
#define HEADS   4
#define DHEAD   32
#define NGRP    16
#define GN_EPS  1e-5f
#define SM_SCALE 0.17677669529663687f   // 1/sqrt(32)

typedef __attribute__((ext_vector_type(16))) __bf16 v16bf;
typedef __attribute__((ext_vector_type(8)))  __bf16 bf16x8;
typedef __attribute__((ext_vector_type(8)))  float  v8f;

// D = A(16x32 bf16) * B(32x16 bf16) + C(16x16 f32)
__device__ __forceinline__ v8f wmma_bf16(v16bf a, v16bf b, v8f c) {
  return __builtin_amdgcn_wmma_f32_16x16x32_bf16(
      /*neg_a=*/false, a, /*neg_b=*/false, b,
      /*c_mod=*/(short)0, c, /*reuse_a=*/false, /*reuse_b=*/false);
}

// ---- fragment loaders (A layout per CDNA5 ISA 7.12.2) ----------------------
// A fragment, 16-bit: lane holds row m=l%16; element e -> k = 8h + (e&7) + 16*(e>>3)
__device__ __forceinline__ v16bf a_frag_bf16(const __bf16* row, int h) {
  bf16x8 u0 = *(const bf16x8*)(row + 8 * h);
  bf16x8 u1 = *(const bf16x8*)(row + 16 + 8 * h);
  v16bf a;
#pragma unroll
  for (int e = 0; e < 8; ++e) { a[e] = u0[e]; a[8 + e] = u1[e]; }
  return a;
}
// B fragment, 16-bit: lane holds col n=l%16; element e -> k = 8h + e (callers pass p at k = 16h)
__device__ __forceinline__ v16bf b_frag_bf16(const __bf16* p) {
  const int hh = (threadIdx.x & 31) >> 4;
  bf16x8 u0 = *(const bf16x8*)(p - 8 * hh);
  bf16x8 u1 = *(const bf16x8*)(p + 8 - 8 * hh);
  v16bf b;
#pragma unroll
  for (int e = 0; e < 8; ++e) { b[e] = u0[e]; b[8 + e] = u1[e]; }
  return b;
}

// reductions across the 16-lane half-group (rows of C-layout live there)
__device__ __forceinline__ float rmax16(float v) {
#pragma unroll
  for (int m = 1; m < 16; m <<= 1) v = fmaxf(v, __shfl_xor(v, m, 32));
  return v;
}
__device__ __forceinline__ float rsum16(float v) {
#pragma unroll
  for (int m = 1; m < 16; m <<= 1) v += __shfl_xor(v, m, 32);
  return v;
}

// ---------------------------------------------------------------------------
// Kernel 0: one-shot fp32 -> bf16 weight conversion (w_qkv 384x128, w_out 128x128)
// ---------------------------------------------------------------------------
__global__ __launch_bounds__(256) void cvt_w_kernel(const float* __restrict__ w_qkv,
                                                    const float* __restrict__ w_out,
                                                    __bf16* __restrict__ wq,
                                                    __bf16* __restrict__ wo) {
  const int i = blockIdx.x * 256 + threadIdx.x;
  if (i < 3 * CCH * CCH) wq[i] = (__bf16)w_qkv[i];
  if (i < CCH * CCH)     wo[i] = (__bf16)w_out[i];
}

// ---------------------------------------------------------------------------
// Kernel 1: GroupNorm statistics. grid = BATCH*NGRP blocks, 256 threads.
// stats[bg*2] = mean, stats[bg*2+1] = rstd
// ---------------------------------------------------------------------------
__global__ __launch_bounds__(256) void gn_stats_kernel(const float* __restrict__ x,
                                                       float* __restrict__ stats) {
  const int bg = blockIdx.x;                 // 0..31
  const int b = bg >> 4, g = bg & 15;
  const float* base = x + ((size_t)b * CCH + (size_t)g * (CCH / NGRP)) * LTOK;
  const int N = (CCH / NGRP) * LTOK;         // 32768
  float s = 0.f, ss = 0.f;
  for (int i = threadIdx.x; i < N; i += blockDim.x) {
    float v = base[i];
    s += v; ss += v * v;
  }
#pragma unroll
  for (int m = 1; m < 32; m <<= 1) { s += __shfl_xor(s, m, 32); ss += __shfl_xor(ss, m, 32); }
  __shared__ float sh[2][8];
  const int wid = threadIdx.x >> 5, lid = threadIdx.x & 31;
  if (lid == 0) { sh[0][wid] = s; sh[1][wid] = ss; }
  __syncthreads();
  if (threadIdx.x == 0) {
    float S = 0.f, SS = 0.f;
#pragma unroll
    for (int w = 0; w < 8; ++w) { S += sh[0][w]; SS += sh[1][w]; }
    float mean = S / (float)N;
    float var  = SS / (float)N - mean * mean;
    stats[bg * 2 + 0] = mean;
    stats[bg * 2 + 1] = rsqrtf(var + GN_EPS);
  }
}

// ---------------------------------------------------------------------------
// Kernel 2: GN-normalize + QKV projection (WMMA GEMM 384x128 @ 128x16).
// grid = BATCH * (LTOK/16) blocks, 256 threads (8 waves, 3 M-tiles each).
// Q,K stored (b,H,L,dh) bf16 ; V stored transposed (b,H,dh,L) bf16.
// Store routing (section / head / dh-base) is wave-uniform per M-tile.
// ---------------------------------------------------------------------------
__global__ __launch_bounds__(256) void qkv_kernel(const float* __restrict__ x,
                                                  const float* __restrict__ gn_w,
                                                  const float* __restrict__ gn_b,
                                                  const __bf16* __restrict__ wq,
                                                  const float* __restrict__ stats,
                                                  __bf16* __restrict__ qbuf,
                                                  __bf16* __restrict__ kbuf,
                                                  __bf16* __restrict__ vtbuf) {
  __shared__ __align__(16) __bf16 xn[16][CCH + 8];   // token-major normalized tile
  const int b    = blockIdx.x >> 8;                  // /256
  const int tcol = blockIdx.x & 255;                 // token tile

  // cooperative load+normalize: thread -> (channel, 8 tokens)
  {
    const int ch   = threadIdx.x >> 1;
    const int htok = threadIdx.x & 1;
    const float* src = x + ((size_t)b * CCH + ch) * LTOK + tcol * 16 + htok * 8;
    const int g = ch >> 3;
    const float mean = stats[(b * NGRP + g) * 2 + 0];
    const float rstd = stats[(b * NGRP + g) * 2 + 1];
    const float w = gn_w[ch], bb = gn_b[ch];
    float4 u0 = ((const float4*)src)[0];
    float4 u1 = ((const float4*)src)[1];
    float t[8] = {u0.x, u0.y, u0.z, u0.w, u1.x, u1.y, u1.z, u1.w};
#pragma unroll
    for (int j = 0; j < 8; ++j)
      xn[htok * 8 + j][ch] = (__bf16)((t[j] - mean) * rstd * w + bb);
  }
  __syncthreads();

  const int lane = threadIdx.x & 31;
  const int wv   = threadIdx.x >> 5;
  const int n    = lane & 15;          // token col / A row
  const int h    = lane >> 4;

#pragma unroll
  for (int t = 0; t < 3; ++t) {
    const int mt = wv * 3 + t;         // 0..23 output-channel tile
    v8f acc = {};
#pragma unroll
    for (int s = 0; s < 4; ++s) {      // K = 128 in 4 steps of 32
      v16bf a  = a_frag_bf16(wq + (size_t)(mt * 16 + n) * CCH + 32 * s, h);
      v16bf bB = b_frag_bf16(&xn[n][32 * s + 16 * h]);
      acc = wmma_bf16(a, bB, acc);
    }
    // --- store: all routing terms are wave-uniform (16-ch tile never crosses
    //     a 128-ch section or 32-ch head boundary) -> single scalar branch ---
    const int token = tcol * 16 + n;
    const int sec = mt >> 3;                 // 0=Q 1=K 2=V   (uniform)
    const int m7  = mt & 7;
    const int hd  = m7 >> 1;                 // head          (uniform)
    const int dhb = (m7 & 1) * 16 + 8 * h;   // dh base
    const size_t bh = (size_t)b * HEADS + hd;
    if (sec == 2) {
      __bf16* dst = vtbuf + (bh * DHEAD + dhb) * LTOK + token;
#pragma unroll
      for (int r = 0; r < 8; ++r) dst[(size_t)r * LTOK] = (__bf16)acc[r];
    } else {
      __bf16* dst = (sec == 0 ? qbuf : kbuf) + (bh * LTOK + token) * DHEAD + dhb;
#pragma unroll
      for (int r = 0; r < 8; ++r) dst[r] = (__bf16)acc[r];
    }
  }
}

// ---------------------------------------------------------------------------
// Kernel 3: flash attention. grid = BATCH*HEADS*(LTOK/128), 256 threads.
// Each wave owns one 16-row Q tile; streams KV in 64-token chunks
// (4 S-WMMAs + 4 PV-WMMAs per chunk -> softmax VALU amortized 2x).
// ---------------------------------------------------------------------------
__global__ __launch_bounds__(256) void attn_kernel(const __bf16* __restrict__ qbuf,
                                                   const __bf16* __restrict__ kbuf,
                                                   const __bf16* __restrict__ vtbuf,
                                                   __bf16* __restrict__ obuf) {
  __shared__ __align__(16) __bf16 pb[8][16][72];   // per-wave P repack (16x64 + pad)

  const int bh = blockIdx.x >> 5;                  // batch*head
  const int rt = blockIdx.x & 31;                  // 128-row super-tile
  const int lane = threadIdx.x & 31;
  const int wv   = threadIdx.x >> 5;
  const int n    = lane & 15;
  const int h    = lane >> 4;
  const int m0   = rt * 128 + wv * 16;

  const __bf16* qb = qbuf  + (size_t)bh * LTOK * DHEAD;
  const __bf16* kb = kbuf  + (size_t)bh * LTOK * DHEAD;
  const __bf16* vt = vtbuf + (size_t)bh * DHEAD * LTOK;

  const v16bf aq = a_frag_bf16(qb + (size_t)(m0 + n) * DHEAD, h);

  float mi[8], li[8];
  v8f o0 = {}, o1 = {};
#pragma unroll
  for (int r = 0; r < 8; ++r) { mi[r] = -3.0e38f; li[r] = 0.f; }

  for (int kc = 0; kc < LTOK / 64; ++kc) {
    const int col0 = kc * 64;
    if (kc + 1 < LTOK / 64) {   // speculative prefetch of next KV chunk -> global_prefetch_b8
      __builtin_prefetch(kb + (size_t)(col0 + 64 + n * 2) * DHEAD, 0, 1);
      __builtin_prefetch(vt + (size_t)n * LTOK + col0 + 64, 0, 1);
      __builtin_prefetch(vt + (size_t)(16 + n) * LTOK + col0 + 64, 0, 1);
    }
    // S = Q * K^T : four 16x16 tiles (K-dim = dh = 32)
    v8f z = {};
    v8f s0, s1, s2, s3;
    {
      v16bf bk0 = b_frag_bf16(kb + (size_t)(col0 + n) * DHEAD + 16 * h);
      v16bf bk1 = b_frag_bf16(kb + (size_t)(col0 + 16 + n) * DHEAD + 16 * h);
      v16bf bk2 = b_frag_bf16(kb + (size_t)(col0 + 32 + n) * DHEAD + 16 * h);
      v16bf bk3 = b_frag_bf16(kb + (size_t)(col0 + 48 + n) * DHEAD + 16 * h);
      s0 = wmma_bf16(aq, bk0, z);
      s1 = wmma_bf16(aq, bk1, z);
      s2 = wmma_bf16(aq, bk2, z);
      s3 = wmma_bf16(aq, bk3, z);
    }

    // online softmax per row (row r+8h lives across the 16-lane half group)
#pragma unroll
    for (int r = 0; r < 8; ++r) {
      float x0 = s0[r] * SM_SCALE;
      float x1 = s1[r] * SM_SCALE;
      float x2 = s2[r] * SM_SCALE;
      float x3 = s3[r] * SM_SCALE;
      float cm = rmax16(fmaxf(fmaxf(x0, x1), fmaxf(x2, x3)));
      float mn = fmaxf(mi[r], cm);
      float al = __expf(mi[r] - mn);
      float e0 = __expf(x0 - mn);
      float e1 = __expf(x1 - mn);
      float e2 = __expf(x2 - mn);
      float e3 = __expf(x3 - mn);
      li[r] = li[r] * al + rsum16((e0 + e1) + (e2 + e3));
      mi[r] = mn;
      o0[r] *= al;
      o1[r] *= al;
      __bf16* prow = &pb[wv][r + 8 * h][0];      // C-layout -> LDS
      prow[n]      = (__bf16)e0;
      prow[16 + n] = (__bf16)e1;
      prow[32 + n] = (__bf16)e2;
      prow[48 + n] = (__bf16)e3;
    }
    // repack P (16x64) into two A fragments (same wave: LDS ops stay in order)
    v16bf ap0 = a_frag_bf16(&pb[wv][n][0],  h);
    v16bf ap1 = a_frag_bf16(&pb[wv][n][32], h);
    // O += P * V  (V transposed so B fragments are contiguous 32B per lane)
    v16bf bv00 = b_frag_bf16(vt + (size_t)n * LTOK + col0 + 16 * h);
    v16bf bv01 = b_frag_bf16(vt + (size_t)n * LTOK + col0 + 32 + 16 * h);
    v16bf bv10 = b_frag_bf16(vt + (size_t)(16 + n) * LTOK + col0 + 16 * h);
    v16bf bv11 = b_frag_bf16(vt + (size_t)(16 + n) * LTOK + col0 + 32 + 16 * h);
    o0 = wmma_bf16(ap0, bv00, o0);
    o0 = wmma_bf16(ap1, bv01, o0);
    o1 = wmma_bf16(ap0, bv10, o1);
    o1 = wmma_bf16(ap1, bv11, o1);
  }

  // finalize: divide by row sum, store O token-major (b, L, 128ch) bf16
  const int bb = bh >> 2, hd = bh & 3;
#pragma unroll
  for (int r = 0; r < 8; ++r) {
    const float inv = 1.0f / li[r];
    const int row = m0 + r + 8 * h;
    __bf16* dst = obuf + ((size_t)bb * LTOK + row) * CCH + hd * DHEAD;
    dst[n]      = (__bf16)(o0[r] * inv);
    dst[16 + n] = (__bf16)(o1[r] * inv);
  }
}

// ---------------------------------------------------------------------------
// Kernel 4: output projection + bias + residual. grid = BATCH*(LTOK/16), 256 thr.
// 8 waves -> 8 M-tiles (128 output channels).
// ---------------------------------------------------------------------------
__global__ __launch_bounds__(256) void outproj_kernel(const __bf16* __restrict__ obuf,
                                                      const __bf16* __restrict__ wo,
                                                      const float* __restrict__ b_out,
                                                      const float* __restrict__ x,
                                                      float* __restrict__ out) {
  const int b    = blockIdx.x >> 8;
  const int tcol = blockIdx.x & 255;
  const int lane = threadIdx.x & 31;
  const int wv   = threadIdx.x >> 5;   // M tile
  const int n    = lane & 15;
  const int h    = lane >> 4;

  v8f acc = {};
#pragma unroll
  for (int s = 0; s < 4; ++s) {
    v16bf a  = a_frag_bf16(wo + (size_t)(wv * 16 + n) * CCH + 32 * s, h);
    v16bf bB = b_frag_bf16(obuf + ((size_t)b * LTOK + tcol * 16 + n) * CCH + 32 * s + 16 * h);
    acc = wmma_bf16(a, bB, acc);
  }
  const int token = tcol * 16 + n;
  const int chb = wv * 16 + 8 * h;
  const float* xs = x   + ((size_t)b * CCH + chb) * LTOK + token;
  float*      os = out + ((size_t)b * CCH + chb) * LTOK + token;
#pragma unroll
  for (int r = 0; r < 8; ++r)
    os[(size_t)r * LTOK] = acc[r] + b_out[chb + r] + xs[(size_t)r * LTOK];
}

// ---------------------------------------------------------------------------
// Launch
// ---------------------------------------------------------------------------
extern "C" void kernel_launch(void* const* d_in, const int* in_sizes, int n_in,
                              void* d_out, int out_size, void* d_ws, size_t ws_size,
                              hipStream_t stream) {
  (void)in_sizes; (void)n_in; (void)out_size; (void)ws_size;
  const float* x     = (const float*)d_in[0];
  const float* gn_w  = (const float*)d_in[1];
  const float* gn_b  = (const float*)d_in[2];
  const float* w_qkv = (const float*)d_in[3];
  const float* w_out = (const float*)d_in[4];
  const float* b_out = (const float*)d_in[5];
  float* out = (float*)d_out;

  char* ws = (char*)d_ws;
  const size_t QBYTES = (size_t)BATCH * HEADS * LTOK * DHEAD * sizeof(__bf16); // 2 MB
  const size_t OBYTES = (size_t)BATCH * LTOK * CCH * sizeof(__bf16);           // 2 MB
  float*  stats = (float*)ws;                               // 256 B
  __bf16* qbuf  = (__bf16*)(ws + 256);
  __bf16* kbuf  = (__bf16*)(ws + 256 + QBYTES);
  __bf16* vtbuf = (__bf16*)(ws + 256 + 2 * QBYTES);
  __bf16* obuf  = (__bf16*)(ws + 256 + 3 * QBYTES);
  __bf16* wq    = (__bf16*)(ws + 256 + 3 * QBYTES + OBYTES);
  __bf16* wo    = (__bf16*)(ws + 256 + 3 * QBYTES + OBYTES +
                            (size_t)3 * CCH * CCH * sizeof(__bf16));

  cvt_w_kernel<<<(3 * CCH * CCH + 255) / 256, 256, 0, stream>>>(w_qkv, w_out, wq, wo);
  gn_stats_kernel<<<BATCH * NGRP, 256, 0, stream>>>(x, stats);
  qkv_kernel<<<BATCH * (LTOK / 16), 256, 0, stream>>>(x, gn_w, gn_b, wq, stats,
                                                      qbuf, kbuf, vtbuf);
  attn_kernel<<<BATCH * HEADS * (LTOK / 128), 256, 0, stream>>>(qbuf, kbuf, vtbuf, obuf);
  outproj_kernel<<<BATCH * (LTOK / 16), 256, 0, stream>>>(obuf, wo, b_out, x, out);
}
